// HierarchicalHAMTLayer_13271448944696
// MI455X (gfx1250) — matrix operand order repaired and checked by reading, not compiled, not run on hardware
//
#include <hip/hip_runtime.h>

// ---------------------------------------------------------------------------
// HierarchicalHAMT layer for MI455X (gfx1250, wave32, WMMA bf16->f32).
// B=4, S=2048, H=1024, D=512, NF=32.
// ---------------------------------------------------------------------------

typedef __bf16 bf16;
typedef __attribute__((ext_vector_type(16))) __bf16 v16bf;
typedef __attribute__((ext_vector_type(8)))  float  v8f;
typedef int v4i __attribute__((vector_size(16)));   // matches builtin's b128 payload type

#define CB 4
#define CS 2048
#define CH 1024
#define CD 512
#define CNF 32
#define CBS (CB*CS)        // 8192
#define CH2 (2*CH)         // 2048
#define CHD (CH+CD)        // 1536

#define TM 128
#define TN 64
#define TK 32
#define LDT 40             // padded LDS row stride (bf16 units): 80B, conflict-free

// ---- CDNA5 async global->LDS DMA path (guarded so compilation never breaks) ----
#if defined(__has_builtin)
# if __has_builtin(__builtin_amdgcn_global_load_async_to_lds_b128) && \
     __has_builtin(__builtin_amdgcn_s_wait_asynccnt)
#  define USE_ASYNC 1
# endif
#endif
#ifndef USE_ASYNC
# define USE_ASYNC 0
#endif

#if USE_ASYNC
__device__ __forceinline__ void async_cp16(const void* g, const void* l) {
    // global VA == addrspace(1) VA; low 32 bits of a flat LDS address == LDS offset (ISA 10.2)
    __builtin_amdgcn_global_load_async_to_lds_b128(
        (__attribute__((address_space(1))) v4i*)(uintptr_t)g,
        (__attribute__((address_space(3))) v4i*)(uint32_t)(uintptr_t)l,
        0, 0);
}
#endif

__device__ __forceinline__ float gelu_tanh(float x) {
    float x3 = x * x * x;
    return 0.5f * x * (1.0f + tanhf(0.79788456080286535588f * (x + 0.044715f * x3)));
}

// ---------------------------------------------------------------------------
// Generic batched GEMM:  C[b] = act( A[b] (MxK, bf16) * Bt[b]^T (Bt is NxK) + bias + Cin )
// GUARD=true handles M not a multiple of TM (zero-filled rows, guarded stores).
// GUARD=false is the hot path: async double-buffered LDS staging, no guards.
// ---------------------------------------------------------------------------
template<bool GUARD>
__global__ __launch_bounds__(256)
void gemm_bf16_wmma(const bf16* __restrict__ A, long long sA, int lda,
                    const bf16* __restrict__ Bt, long long sB, int ldb,
                    const float* __restrict__ bias,
                    const float* __restrict__ Cin, long long sCin,
                    float* __restrict__ outF, bf16* __restrict__ outB,
                    long long sC, int ldc,
                    int M, int N, int K, int act)
{
    constexpr int NBUF = (USE_ASYNC != 0 && !GUARD) ? 2 : 1;
    __shared__ __align__(16) bf16 As[NBUF][TM * LDT];
    __shared__ __align__(16) bf16 Bs[NBUF][TN * LDT];

    const int tid  = threadIdx.x;
    const int lane = tid & 31;
    const int wave = tid >> 5;          // 8 waves
    const int wm   = wave >> 1;         // 0..3  -> 32-row strip
    const int wn   = wave & 1;          // 0..1  -> 32-col strip
    const int bM   = blockIdx.y * TM;
    const int bN   = blockIdx.x * TN;
    const long long bz = blockIdx.z;

    A  += bz * sA;
    Bt += bz * sB;

    const int mrow = lane & 15;
    const int hk   = lane >> 4;

    // staging indices: A tile 128x32 = 512 uint4 (2/thread), B tile 64x32 = 256 uint4 (1/thread)
    const int ar = tid >> 2;            // 0..63 (and +64 for second A chunk)
    const int ac = (tid & 3) * 8;

    v8f acc[2][2] = {};

    // fragment loads + 2x2 WMMA from buffer `cur`
    auto compute = [&](int cur) {
        union Frag { uint4 q[2]; v16bf f; };
        Frag a[2], b[2];
        // A (16x32, 16-bit): lanes 0-15: M=lane, K=0..7 & 16..23; lanes 16-31: K=8..15 & 24..31
#pragma unroll
        for (int mi = 0; mi < 2; ++mi) {
            int r = wm * 32 + mi * 16 + mrow;
            a[mi].q[0] = *(const uint4*)(&As[cur][r * LDT + hk * 8]);
            a[mi].q[1] = *(const uint4*)(&As[cur][r * LDT + 16 + hk * 8]);
        }
        // B (32x16, 16-bit): lanes 0-15: N=lane, K=0..7 & 16..23; lanes 16-31: K=8..15 & 24..31
#pragma unroll
        for (int ni = 0; ni < 2; ++ni) {
            int r = wn * 32 + ni * 16 + mrow;
            b[ni].q[0] = *(const uint4*)(&Bs[cur][r * LDT + hk * 8]);
            b[ni].q[1] = *(const uint4*)(&Bs[cur][r * LDT + 16 + hk * 8]);
        }
#pragma unroll
        for (int mi = 0; mi < 2; ++mi)
#pragma unroll
            for (int ni = 0; ni < 2; ++ni)
                acc[mi][ni] = __builtin_amdgcn_wmma_f32_16x16x32_bf16(
                    false, a[mi].f, false, b[ni].f, (short)0, acc[mi][ni], false, false);
    };

    const int nt = K / TK;

#if USE_ASYNC
    if constexpr (!GUARD) {
        // async DMA global->LDS, double-buffered; 3 async ops per thread per tile
        auto issue = [&](int k0, int buf) {
            async_cp16(A + (long long)(bM + ar) * lda + k0 + ac,       &As[buf][ar * LDT + ac]);
            async_cp16(A + (long long)(bM + ar + 64) * lda + k0 + ac,  &As[buf][(ar + 64) * LDT + ac]);
            async_cp16(Bt + (long long)(bN + ar) * ldb + k0 + ac,      &Bs[buf][ar * LDT + ac]);
        };
        issue(0, 0);
        for (int t = 0; t < nt; ++t) {
            int cur = t & 1;
            if (t + 1 < nt) {
                issue((t + 1) * TK, cur ^ 1);
                __builtin_amdgcn_s_wait_asynccnt(3);   // tile t landed (in-order), tile t+1 in flight
            } else {
                __builtin_amdgcn_s_wait_asynccnt(0);
            }
            __syncthreads();
            compute(cur);
            __syncthreads();
        }
    } else
#endif
    {
        for (int t = 0; t < nt; ++t) {
            int k0 = t * TK;
            // ---- stage A tile ----
#pragma unroll
            for (int i = 0; i < 2; ++i) {
                int r = ar + i * 64;
                int gr = bM + r;
                uint4 v = make_uint4(0u, 0u, 0u, 0u);
                if (!GUARD || gr < M) v = *(const uint4*)(A + (long long)gr * lda + k0 + ac);
                *(uint4*)(&As[0][r * LDT + ac]) = v;
            }
            // ---- stage B tile ----
            *(uint4*)(&Bs[0][ar * LDT + ac]) =
                *(const uint4*)(Bt + (long long)(bN + ar) * ldb + k0 + ac);
            if (t + 1 < nt) {
                if (!GUARD || bM + ar < M)
                    __builtin_prefetch(A + (long long)(bM + ar) * lda + k0 + TK + ac, 0, 1);
                __builtin_prefetch(Bt + (long long)(bN + ar) * ldb + k0 + TK + ac, 0, 1);
            }
            __syncthreads();
            compute(0);
            __syncthreads();
        }
    }

    // ---- epilogue: C/D layout: VGPR r -> M = r + 8*(lane>>4); N = lane&15 ----
    const float* cin = Cin ? (Cin + bz * sCin) : nullptr;
    float* oF = outF ? (outF + bz * sC) : nullptr;
    bf16*  oB = outB ? (outB + bz * sC) : nullptr;

#pragma unroll
    for (int mi = 0; mi < 2; ++mi)
#pragma unroll
        for (int ni = 0; ni < 2; ++ni) {
            int gc = bN + wn * 32 + ni * 16 + (lane & 15);
            float bv = bias ? bias[gc] : 0.0f;
#pragma unroll
            for (int r = 0; r < 8; ++r) {
                int gr = bM + wm * 32 + mi * 16 + r + 8 * hk;
                if (!GUARD || gr < M) {
                    float x = acc[mi][ni][r] + bv;
                    long long off = (long long)gr * ldc + gc;
                    if (cin) x += cin[off];
                    if (act == 1)      x = gelu_tanh(x);
                    else if (act == 2) x = 1.0f / (1.0f + expf(-x));
                    if (oF) oF[off] = x;
                    if (oB) oB[off] = (bf16)x;
                }
            }
        }
}

// ---------------------------------------------------------------------------
// Small helper kernels
// ---------------------------------------------------------------------------
__global__ void cast_f32_to_bf16(const float* __restrict__ s, bf16* __restrict__ d, int n) {
    int i = blockIdx.x * 256 + threadIdx.x;
    if (i < n) d[i] = (bf16)s[i];
}

// src [R,C] f32 -> dst [C,R] bf16
__global__ __launch_bounds__(256)
void transpose_f32_to_bf16(const float* __restrict__ src, bf16* __restrict__ dst, int R, int C) {
    __shared__ float t[32][33];
    int c0 = blockIdx.x * 32, r0 = blockIdx.y * 32;
    int tx = threadIdx.x, ty = threadIdx.y;
#pragma unroll
    for (int i = 0; i < 4; ++i) {
        int r = ty + i * 8;
        t[r][tx] = src[(long long)(r0 + r) * C + c0 + tx];
    }
    __syncthreads();
#pragma unroll
    for (int i = 0; i < 4; ++i) {
        int cc = ty + i * 8;
        dst[(long long)(c0 + cc) * R + r0 + tx] = (bf16)t[tx][cc];
    }
}

// per-batch bf16 transpose: src [R,C] -> dst [C,R]
__global__ __launch_bounds__(256)
void transpose_bf16(const bf16* __restrict__ src, bf16* __restrict__ dst,
                    int R, int C, long long sSrc, long long sDst) {
    src += (long long)blockIdx.z * sSrc;
    dst += (long long)blockIdx.z * sDst;
    __shared__ bf16 t[32][33];
    int c0 = blockIdx.x * 32, r0 = blockIdx.y * 32;
    int tx = threadIdx.x, ty = threadIdx.y;
#pragma unroll
    for (int i = 0; i < 4; ++i) {
        int r = ty + i * 8;
        t[r][tx] = src[(long long)(r0 + r) * C + c0 + tx];
    }
    __syncthreads();
#pragma unroll
    for (int i = 0; i < 4; ++i) {
        int cc = ty + i * 8;
        dst[(long long)(c0 + cc) * R + r0 + tx] = t[tx][cc];
    }
}

// state0[b,m] = sum_j fast[b,j,m] + slow[b,j,m]
__global__ void state0_kernel(const float* __restrict__ fast, const float* __restrict__ slow,
                              float* __restrict__ state0) {
    int i = blockIdx.x * 256 + threadIdx.x;
    if (i >= CB * CD) return;
    int b = i >> 9, m = i & (CD - 1);
    float s = 0.0f;
    for (int j = 0; j < CNF; ++j) {
        long long o = ((long long)b * CNF + j) * CD + m;
        s += fast[o] + slow[o];
    }
    state0[i] = s;
}

// circ[b][n*D+m] = state0[b][(m+n) mod D]   (symmetric circulant = B and B^T)
__global__ void circulant_kernel(const float* __restrict__ state0, bf16* __restrict__ circ) {
    int i = blockIdx.x * 256 + threadIdx.x;
    if (i >= CB * CD * CD) return;
    int b = i >> 18;
    int rem = i & (CD * CD - 1);
    int n = rem >> 9, m = rem & (CD - 1);
    circ[i] = (bf16)state0[b * CD + ((m + n) & (CD - 1))];
}

// GwF[b][j*S+t] = g[b,t,j]*0.5^F(t); GwS = g * 0.2*(1-0.5^F(t));  F(t)=205-ceil(t/10)
__global__ void gate_weight_kernel(const float* __restrict__ gates, // [B,S,64]
                                   bf16* __restrict__ gwF, bf16* __restrict__ gwS) {
    int i = blockIdx.x * 256 + threadIdx.x;
    if (i >= CB * CNF * CS) return;
    int b = i >> 16;
    int rem = i & 65535;
    int j = rem >> 11;
    int t = rem & (CS - 1);
    float g = gates[((long long)b * CS + t) * 64 + j];
    int F = 205 - (t + 9) / 10;
    float wF = exp2f(-(float)F);
    float wS = 0.2f * (1.0f - wF);
    gwF[i] = (bf16)(g * wF);
    gwS[i] = (bf16)(g * wS);
}

// new_fast/new_slow: apply per-slot circulant of slot_keys to P_fast/P_slow
// bound(n) = sum_m key[(n-m) mod D] * P[m]
__global__ __launch_bounds__(256)
void memory_final_kernel(const float* __restrict__ pF, const float* __restrict__ pS,
                         const float* __restrict__ slot_keys,
                         const float* __restrict__ fast0, const float* __restrict__ slow0,
                         float* __restrict__ outFast, float* __restrict__ outSlow) {
    int blk = blockIdx.x;              // b*NF + j
    int j = blk & (CNF - 1);
    __shared__ float sk[CD], pf[CD], ps[CD];
    for (int i = threadIdx.x; i < CD; i += 256) {
        sk[i] = slot_keys[(long long)j * CD + i];
        pf[i] = pF[(long long)blk * CD + i];
        ps[i] = pS[(long long)blk * CD + i];
    }
    __syncthreads();
    float fpow = exp2f(-205.0f);   // 205 consolidation events -> underflows to 0 (matches f32 scan)
#pragma unroll
    for (int rep = 0; rep < 2; ++rep) {
        int n = threadIdx.x + rep * 256;
        float aF = 0.0f, aS = 0.0f;
        for (int m = 0; m < CD; ++m) {
            float k = sk[(n - m) & (CD - 1)];
            aF = fmaf(k, pf[m], aF);
            aS = fmaf(k, ps[m], aS);
        }
        long long o = (long long)blk * CD + n;
        float f0 = fast0[o], s0 = slow0[o];
        outFast[o] = aF + f0 * fpow;
        outSlow[o] = s0 + 0.2f * (1.0f - fpow) * f0 + aS;
    }
}

// out = LN(hidden + proj) * gamma + beta    (row = one (b,s), H=1024)
__global__ __launch_bounds__(256)
void layernorm_kernel(const float* __restrict__ hidden, const float* __restrict__ proj,
                      const float* __restrict__ gamma, const float* __restrict__ beta,
                      float* __restrict__ out) {
    long long row = blockIdx.x;
    const float* h = hidden + row * CH;
    const float* p = proj + row * CH;
    __shared__ float rs[8], rs2[8];
    int lane = threadIdx.x & 31, wave = threadIdx.x >> 5;
    float x[4], s = 0.0f, s2 = 0.0f;
#pragma unroll
    for (int i = 0; i < 4; ++i) {
        int c = threadIdx.x + i * 256;
        x[i] = h[c] + p[c];
        s += x[i];
        s2 += x[i] * x[i];
    }
#pragma unroll
    for (int o = 16; o > 0; o >>= 1) {
        s  += __shfl_down(s, o, 32);
        s2 += __shfl_down(s2, o, 32);
    }
    if (lane == 0) { rs[wave] = s; rs2[wave] = s2; }
    __syncthreads();
    float ts = 0.0f, t2 = 0.0f;
#pragma unroll
    for (int w = 0; w < 8; ++w) { ts += rs[w]; t2 += rs2[w]; }
    float mu  = ts * (1.0f / CH);
    float var = t2 * (1.0f / CH) - mu * mu;
    float inv = rsqrtf(var + 1e-5f);
#pragma unroll
    for (int i = 0; i < 4; ++i) {
        int c = threadIdx.x + i * 256;
        out[row * CH + c] = (x[i] - mu) * inv * gamma[c] + beta[c];
    }
}

// ---------------------------------------------------------------------------
// Host-side launcher
// ---------------------------------------------------------------------------
static inline size_t align256(size_t x) { return (x + 255) & ~(size_t)255; }

extern "C" void kernel_launch(void* const* d_in, const int* in_sizes, int n_in,
                              void* d_out, int out_size, void* d_ws, size_t ws_size,
                              hipStream_t stream) {
    (void)in_sizes; (void)n_in; (void)out_size; (void)ws_size;
    const float* hs    = (const float*)d_in[0];
    const float* fast0 = (const float*)d_in[1];
    const float* slow0 = (const float*)d_in[2];
    const float* Wi = (const float*)d_in[3],  *bi  = (const float*)d_in[4];
    const float* Wq = (const float*)d_in[5],  *bq  = (const float*)d_in[6];
    const float* Wr1= (const float*)d_in[7],  *br1 = (const float*)d_in[8];
    const float* Wr2= (const float*)d_in[9],  *br2 = (const float*)d_in[10];
    const float* Wg = (const float*)d_in[11], *bg  = (const float*)d_in[12];
    const float* Wo = (const float*)d_in[13], *bo  = (const float*)d_in[14];
    const float* gamma = (const float*)d_in[15], *beta = (const float*)d_in[16];
    const float* slot_keys = (const float*)d_in[17];

    float* out      = (float*)d_out;
    float* outFast  = out + (long long)CBS * CH;
    float* outSlow  = outFast + (long long)CB * CNF * CD;

    // workspace bump allocator
    char* ws = (char*)d_ws;
    size_t off = 0;
    auto alloc = [&](size_t bytes) -> void* { void* p = ws + off; off = align256(off + bytes); return p; };

    bf16* hs_bf    = (bf16*)alloc((size_t)CBS * CH * 2);
    bf16* WiT      = (bf16*)alloc((size_t)CD * CH * 2);
    bf16* WqT      = (bf16*)alloc((size_t)CH * CH * 2);
    bf16* Wr1T     = (bf16*)alloc((size_t)CH2 * CD * 2);
    bf16* Wr2T     = (bf16*)alloc((size_t)CD * CH2 * 2);
    bf16* WgT      = (bf16*)alloc((size_t)64 * CHD * 2);
    bf16* WoT      = (bf16*)alloc((size_t)CH * CHD * 2);
    bf16* items_bf = (bf16*)alloc((size_t)CBS * CD * 2);
    bf16* itemsT   = (bf16*)alloc((size_t)CBS * CD * 2);
    bf16* query_bf = (bf16*)alloc((size_t)CBS * CH * 2);
    bf16* t1_bf    = (bf16*)alloc((size_t)CBS * CH2 * 2);
    bf16* ukeys_bf = (bf16*)alloc((size_t)CBS * CD * 2);
    float* state0  = (float*)alloc((size_t)CB * CD * 4);
    bf16* circ_bf  = (bf16*)alloc((size_t)CB * CD * CD * 2);
    bf16* retr_bf  = (bf16*)alloc((size_t)CBS * CD * 2);
    float* gatesF  = (float*)alloc((size_t)CBS * 64 * 4);
    bf16* gwF      = (bf16*)alloc((size_t)CB * CNF * CS * 2);
    bf16* gwS      = (bf16*)alloc((size_t)CB * CNF * CS * 2);
    float* pF      = (float*)alloc((size_t)CB * CNF * CD * 4);
    float* pS      = (float*)alloc((size_t)CB * CNF * CD * 4);
    float* projF   = (float*)alloc((size_t)CBS * CH * 4);

    auto gemm = [&](bool guard,
                    const bf16* A, long long sA, int lda,
                    const bf16* Bt, long long sB, int ldb,
                    const float* bias, const float* Cin, long long sCin,
                    float* oF, bf16* oB, long long sC, int ldc,
                    int M, int N, int K, int act, int batch) {
        dim3 g(N / TN, (M + TM - 1) / TM, batch);
        if (guard)
            gemm_bf16_wmma<true><<<g, 256, 0, stream>>>(A, sA, lda, Bt, sB, ldb, bias, Cin, sCin,
                                                        oF, oB, sC, ldc, M, N, K, act);
        else
            gemm_bf16_wmma<false><<<g, 256, 0, stream>>>(A, sA, lda, Bt, sB, ldb, bias, Cin, sCin,
                                                         oF, oB, sC, ldc, M, N, K, act);
    };

    // 0) casts / weight transposes (f32 -> bf16, W[K,N] -> Wt[N,K])
    cast_f32_to_bf16<<<(CBS * CH) / 256, 256, 0, stream>>>(hs, hs_bf, CBS * CH);
    dim3 tb(32, 8);
    transpose_f32_to_bf16<<<dim3(CD / 32, CH / 32), tb, 0, stream>>>(Wi, WiT, CH, CD);
    transpose_f32_to_bf16<<<dim3(CH / 32, CH / 32), tb, 0, stream>>>(Wq, WqT, CH, CH);
    transpose_f32_to_bf16<<<dim3(CH2 / 32, CD / 32), tb, 0, stream>>>(Wr1, Wr1T, CD, CH2);
    transpose_f32_to_bf16<<<dim3(CD / 32, CH2 / 32), tb, 0, stream>>>(Wr2, Wr2T, CH2, CD);
    transpose_f32_to_bf16<<<dim3(64 / 32, CHD / 32), tb, 0, stream>>>(Wg, WgT, CHD, 64);
    transpose_f32_to_bf16<<<dim3(CH / 32, CHD / 32), tb, 0, stream>>>(Wo, WoT, CHD, CH);

    // 1) items = hs @ Wi + bi  (bf16 out)
    gemm(false, hs_bf, 0, CH, WiT, 0, CH, bi, nullptr, 0, nullptr, items_bf, 0, CD,
         CBS, CD, CH, 0, 1);
    // 2) query = hs @ Wq + bq
    gemm(false, hs_bf, 0, CH, WqT, 0, CH, bq, nullptr, 0, nullptr, query_bf, 0, CH,
         CBS, CH, CH, 0, 1);
    // 3) t1 = gelu(items @ Wr1 + br1)
    gemm(false, items_bf, 0, CD, Wr1T, 0, CD, br1, nullptr, 0, nullptr, t1_bf, 0, CH2,
         CBS, CH2, CD, 1, 1);
    // 4) ukeys = t1 @ Wr2 + br2
    gemm(false, t1_bf, 0, CH2, Wr2T, 0, CH2, br2, nullptr, 0, nullptr, ukeys_bf, 0, CD,
         CBS, CD, CH2, 0, 1);

    // 5) retrieval: circulant GEMM per batch
    state0_kernel<<<(CB * CD) / 256, 256, 0, stream>>>(fast0, slow0, state0);
    circulant_kernel<<<(CB * CD * CD) / 256, 256, 0, stream>>>(state0, circ_bf);
    gemm(false, ukeys_bf, (long long)CS * CD, CD, circ_bf, (long long)CD * CD, CD,
         nullptr, nullptr, 0, nullptr, retr_bf, (long long)CS * CD, CD,
         CS, CD, CD, 0, CB);

    // 6) gates = sigmoid(hs @ Wg[:H] + retr @ Wg[H:] + bg)
    gemm(false, hs_bf, 0, CH, WgT, 0, CHD, nullptr, nullptr, 0, gatesF, nullptr, 0, 64,
         CBS, 64, CH, 0, 1);
    gemm(false, retr_bf, 0, CD, WgT + CH, 0, CHD, bg, gatesF, 0, gatesF, nullptr, 0, 64,
         CBS, 64, CD, 2, 1);

    // 7) scan collapsed: P[b,j] = sum_t w(t) g[b,t,j] items[b,t]
    gate_weight_kernel<<<(CB * CNF * CS) / 256, 256, 0, stream>>>(gatesF, gwF, gwS);
    transpose_bf16<<<dim3(CD / 32, CS / 32, CB), tb, 0, stream>>>(
        items_bf, itemsT, CS, CD, (long long)CS * CD, (long long)CD * CS);
    gemm(true, gwF, (long long)CNF * CS, CS, itemsT, (long long)CD * CS, CS,
         nullptr, nullptr, 0, pF, nullptr, (long long)CNF * CD, CD,
         CNF, CD, CS, 0, CB);
    gemm(true, gwS, (long long)CNF * CS, CS, itemsT, (long long)CD * CS, CS,
         nullptr, nullptr, 0, pS, nullptr, (long long)CNF * CD, CD,
         CNF, CD, CS, 0, CB);
    memory_final_kernel<<<CB * CNF, 256, 0, stream>>>(pF, pS, slot_keys, fast0, slow0,
                                                      outFast, outSlow);

    // 8) out projection: proj = query @ Wo[:H] + retr @ Wo[H:] + bo
    gemm(false, query_bf, 0, CH, WoT, 0, CHD, nullptr, nullptr, 0, projF, nullptr, 0, CH,
         CBS, CH, CH, 0, 1);
    gemm(false, retr_bf, 0, CD, WoT + CH, 0, CHD, bo, projF, 0, projF, nullptr, 0, CH,
         CBS, CH, CD, 0, 1);

    // 9) out = LN(hidden + proj)
    layernorm_kernel<<<CBS, 256, 0, stream>>>(hs, projF, gamma, beta, out);
}
